// CausalSelfAttention_16389595202207
// MI455X (gfx1250) — matrix operand order repaired and checked by reading, not compiled, not run on hardware
//
#include <hip/hip_runtime.h>
#include <hip/hip_bf16.h>
#include <stdint.h>

typedef __attribute__((ext_vector_type(16))) __bf16 v16bf;
typedef __attribute__((ext_vector_type(8)))  float  v8f;

#define WMMA_BF16(a, b, c) \
  __builtin_amdgcn_wmma_f32_16x16x32_bf16(false, (a), false, (b), (short)0, (c), false, false)

__device__ __forceinline__ unsigned short f2bf(float f) {
  unsigned u = __builtin_bit_cast(unsigned, f);
  u = u + 0x7FFFu + ((u >> 16) & 1u);   // round-to-nearest-even
  return (unsigned short)(u >> 16);
}

struct Frag { union { v16bf v; uint4 u[2]; }; };

// A-fragment (16x32 bf16, row-major source, row stride in elements).
// Lanes 0-15: row m=lane, K chunks {0..7, 16..23}; lanes 16-31: row m=lane-16, K {8..15, 24..31}.
__device__ __forceinline__ v16bf load_a_frag(const unsigned short* base, int stride, int lane) {
  int m   = lane & 15;
  int off = (lane < 16) ? 0 : 8;
  const unsigned short* p = base + (size_t)m * stride + off;
  Frag f;
  f.u[0] = *(const uint4*)(p);
  f.u[1] = *(const uint4*)(p + 16);
  return f.v;
}

// B-fragment (32x16 bf16). Source holds B[k,n] = mem[n*stride + k] (i.e. column n is a
// contiguous row of the source). Lanes 0-15: n=lane, K {0..7, 16..23}; lanes 16-31: n=lane-16, K {8..15, 24..31}.
__device__ __forceinline__ v16bf load_b_frag(const unsigned short* base, int stride, int lane) {
  int n  = lane & 15;
  int kb = (lane < 16) ? 0 : 8;
  const unsigned short* p = base + (size_t)n * stride + kb;
  Frag f;
  f.u[0] = *(const uint4*)(p);
  f.u[1] = *(const uint4*)(p + 16);
  return f.v;
}

// ---------------------------------------------------------------- converts
__global__ void f32_to_bf16_k(const float* __restrict__ in, unsigned short* __restrict__ out, int n) {
  int i = blockIdx.x * blockDim.x + threadIdx.x;
  if (i < n) out[i] = f2bf(in[i]);
}

// ---------------------------------------------------------------- GEMMs: C = A @ W^T + bias
// A: [M,K] bf16 row-major. W: [N,K] bf16 row-major. One wave = 16(M) x 64(N) tile.
// Depth-2 consume-then-refill pipeline (K % 64 == 0, K >= 64): every fragment set is loaded
// a full 4-WMMA window before its consuming WMMAs -> partial s_wait_loadcnt on both halves.
__global__ __launch_bounds__(256) void gemm_bf16out(
    const unsigned short* __restrict__ A, const unsigned short* __restrict__ W,
    const float* __restrict__ bias, unsigned short* __restrict__ out, int K, int N) {
  int wave = (int)((blockIdx.x * blockDim.x + threadIdx.x) >> 5);
  int lane = threadIdx.x & 31;
  int nsup = N >> 6;
  int mt = wave / nsup, ns = wave % nsup;
  int n0 = ns * 64;
  const unsigned short* arow = A + (size_t)mt * 16 * K;
  const unsigned short* wrow = W + (size_t)n0 * K;
  const unsigned short* aPf  = arow + (size_t)(lane & 15) * K + ((lane < 16) ? 0 : 8);

  v8f acc[4] = { {0,0,0,0,0,0,0,0}, {0,0,0,0,0,0,0,0}, {0,0,0,0,0,0,0,0}, {0,0,0,0,0,0,0,0} };
  v16bf a0 = load_a_frag(arow, K, lane);
  v16bf b0[4];
#pragma unroll
  for (int j = 0; j < 4; j++) b0[j] = load_b_frag(wrow + (size_t)j * 16 * K, K, lane);
  v16bf a1 = load_a_frag(arow + 32, K, lane);
  v16bf b1[4];
#pragma unroll
  for (int j = 0; j < 4; j++) b1[j] = load_b_frag(wrow + (size_t)j * 16 * K + 32, K, lane);

  for (int k = 0; k < K; k += 64) {
#pragma unroll
    for (int j = 0; j < 4; j++) acc[j] = WMMA_BF16(a0, b0[j], acc[j]);
    if (k + 64 < K) {                                        // refill buf0 for k+64
      a0 = load_a_frag(arow + k + 64, K, lane);
#pragma unroll
      for (int j = 0; j < 4; j++) b0[j] = load_b_frag(wrow + (size_t)j * 16 * K + k + 64, K, lane);
    }
    if (k + 128 < K) __builtin_prefetch(aPf + k + 128, 0, 1);  // global_prefetch_b8
#pragma unroll
    for (int j = 0; j < 4; j++) acc[j] = WMMA_BF16(a1, b1[j], acc[j]);
    if (k + 96 < K) {                                        // refill buf1 for k+96
      a1 = load_a_frag(arow + k + 96, K, lane);
#pragma unroll
      for (int j = 0; j < 4; j++) b1[j] = load_b_frag(wrow + (size_t)j * 16 * K + k + 96, K, lane);
    }
  }

  int moff = (lane >= 16) ? 8 : 0, nn = lane & 15;
#pragma unroll
  for (int j = 0; j < 4; j++) {
    float bv = bias[n0 + j * 16 + nn];
#pragma unroll
    for (int e = 0; e < 8; e++)
      out[(size_t)(mt * 16 + e + moff) * N + n0 + j * 16 + nn] = f2bf(acc[j][e] + bv);
  }
}

__global__ __launch_bounds__(256) void gemm_f32out(
    const unsigned short* __restrict__ A, const unsigned short* __restrict__ W,
    const float* __restrict__ bias, float* __restrict__ out, int K, int N) {
  int wave = (int)((blockIdx.x * blockDim.x + threadIdx.x) >> 5);
  int lane = threadIdx.x & 31;
  int nsup = N >> 6;
  int mt = wave / nsup, ns = wave % nsup;
  int n0 = ns * 64;
  const unsigned short* arow = A + (size_t)mt * 16 * K;
  const unsigned short* wrow = W + (size_t)n0 * K;
  const unsigned short* aPf  = arow + (size_t)(lane & 15) * K + ((lane < 16) ? 0 : 8);

  v8f acc[4] = { {0,0,0,0,0,0,0,0}, {0,0,0,0,0,0,0,0}, {0,0,0,0,0,0,0,0}, {0,0,0,0,0,0,0,0} };
  v16bf a0 = load_a_frag(arow, K, lane);
  v16bf b0[4];
#pragma unroll
  for (int j = 0; j < 4; j++) b0[j] = load_b_frag(wrow + (size_t)j * 16 * K, K, lane);
  v16bf a1 = load_a_frag(arow + 32, K, lane);
  v16bf b1[4];
#pragma unroll
  for (int j = 0; j < 4; j++) b1[j] = load_b_frag(wrow + (size_t)j * 16 * K + 32, K, lane);

  for (int k = 0; k < K; k += 64) {
#pragma unroll
    for (int j = 0; j < 4; j++) acc[j] = WMMA_BF16(a0, b0[j], acc[j]);
    if (k + 64 < K) {
      a0 = load_a_frag(arow + k + 64, K, lane);
#pragma unroll
      for (int j = 0; j < 4; j++) b0[j] = load_b_frag(wrow + (size_t)j * 16 * K + k + 64, K, lane);
    }
    if (k + 128 < K) __builtin_prefetch(aPf + k + 128, 0, 1);
#pragma unroll
    for (int j = 0; j < 4; j++) acc[j] = WMMA_BF16(a1, b1[j], acc[j]);
    if (k + 96 < K) {
      a1 = load_a_frag(arow + k + 96, K, lane);
#pragma unroll
      for (int j = 0; j < 4; j++) b1[j] = load_b_frag(wrow + (size_t)j * 16 * K + k + 96, K, lane);
    }
  }

  int moff = (lane >= 16) ? 8 : 0, nn = lane & 15;
#pragma unroll
  for (int j = 0; j < 4; j++) {
    float bv = bias[n0 + j * 16 + nn];
#pragma unroll
    for (int e = 0; e < 8; e++)
      out[(size_t)(mt * 16 + e + moff) * N + n0 + j * 16 + nn] = acc[j][e] + bv;
  }
}

// ---------------------------------------------------------------- V transpose: Vt[b,h,d,t]
__global__ void build_vt_k(const unsigned short* __restrict__ qkv, unsigned short* __restrict__ vt) {
  const int T = 2048, C3 = 3072;
  int idx = blockIdx.x * blockDim.x + threadIdx.x;   // over B*H*64*T = 8,388,608
  int t = idx & (T - 1);
  int d = (idx >> 11) & 63;
  int h = (idx >> 17) & 15;
  int b = idx >> 21;
  vt[idx] = qkv[(size_t)(b * T + t) * C3 + 2048 + h * 64 + d];
}

// ---------------------------------------------------------------- fused causal attention
// One wave = 16 query rows of one (b,h). 3 passes over key tiles: max, sum, write+P@V.
// Depth-2 consume-then-refill pipelines in every pass.
__global__ __launch_bounds__(256) void attn_kernel(
    const unsigned short* __restrict__ qkv, const unsigned short* __restrict__ vt,
    float* __restrict__ attnw, unsigned short* __restrict__ oh) {
  __shared__ __align__(16) unsigned short pstage[8][512];   // 16x32 bf16 per wave
  const int T = 2048, C3 = 3072;
  int wid = threadIdx.x >> 5, lane = threadIdx.x & 31;
  int task = blockIdx.x * 8 + wid;            // 8192 tasks total
  int qt = task & 127;
  int h  = (task >> 7) & 15;
  int b  = task >> 11;
  int q0 = qt * 16;

  const unsigned short* qbase = qkv + (size_t)b * T * C3 + h * 64;
  const unsigned short* kbase = qkv + (size_t)b * T * C3 + 1024 + h * 64;
  const unsigned short* vtb   = vt + (size_t)(b * 16 + h) * 64 * T;
  float* aw = attnw + (size_t)(b * 16 + h) * T * T + (size_t)q0 * T;
  unsigned short* pst = &pstage[wid][0];
  const unsigned short* kPf = kbase + (size_t)(lane & 15) * C3;   // 1 cacheline/lane covers a K tile

  v16bf qf0 = load_a_frag(qbase + (size_t)q0 * C3,      C3, lane);
  v16bf qf1 = load_a_frag(qbase + (size_t)q0 * C3 + 32, C3, lane);

  int moff = (lane >= 16) ? 8 : 0, nn = lane & 15;
  const float scale = 0.125f;                  // 1/sqrt(64)

  // ---- pass 1: row max
  float rmax[8];
#pragma unroll
  for (int e = 0; e < 8; e++) rmax[e] = -1e30f;
  {
    auto tile = [&](int kt, v16bf f0, v16bf f1) {
      int k0 = kt * 16;
      v8f s = {0,0,0,0,0,0,0,0};
      s = WMMA_BF16(qf0, f0, s);
      s = WMMA_BF16(qf1, f1, s);
#pragma unroll
      for (int e = 0; e < 8; e++)
        if (k0 + nn <= q0 + e + moff) rmax[e] = fmaxf(rmax[e], s[e] * scale);
    };
    v16bf e0 = load_b_frag(kbase, C3, lane);
    v16bf e1 = load_b_frag(kbase + 32, C3, lane);
    int pl = (qt >= 1) ? 1 : 0;   // prologue buf1 tile (dummy-load tile 0 when qt==0)
    v16bf g0 = load_b_frag(kbase + (size_t)pl * 16 * C3,      C3, lane);
    v16bf g1 = load_b_frag(kbase + (size_t)pl * 16 * C3 + 32, C3, lane);
    int kt = 0;
    while (kt + 1 <= qt) {
      tile(kt, e0, e1);
      if (kt + 2 <= qt) {
        e0 = load_b_frag(kbase + (size_t)(kt + 2) * 16 * C3,      C3, lane);
        e1 = load_b_frag(kbase + (size_t)(kt + 2) * 16 * C3 + 32, C3, lane);
      }
      if (kt + 4 <= qt) __builtin_prefetch(kPf + (size_t)(kt + 4) * 16 * C3, 0, 1);
      tile(kt + 1, g0, g1);
      if (kt + 3 <= qt) {
        g0 = load_b_frag(kbase + (size_t)(kt + 3) * 16 * C3,      C3, lane);
        g1 = load_b_frag(kbase + (size_t)(kt + 3) * 16 * C3 + 32, C3, lane);
      }
      kt += 2;
    }
    if (kt <= qt) tile(kt, e0, e1);
  }
#pragma unroll
  for (int e = 0; e < 8; e++) {
    float v = rmax[e];
#pragma unroll
    for (int m = 1; m < 16; m <<= 1) v = fmaxf(v, __shfl_xor(v, m, 32));
    rmax[e] = v;
  }

  // ---- pass 2: row sum of exp(s - max)
  float rsum[8];
#pragma unroll
  for (int e = 0; e < 8; e++) rsum[e] = 0.f;
  {
    auto tile = [&](int kt, v16bf f0, v16bf f1) {
      int k0 = kt * 16;
      v8f s = {0,0,0,0,0,0,0,0};
      s = WMMA_BF16(qf0, f0, s);
      s = WMMA_BF16(qf1, f1, s);
#pragma unroll
      for (int e = 0; e < 8; e++)
        if (k0 + nn <= q0 + e + moff) rsum[e] += __expf(s[e] * scale - rmax[e]);
    };
    v16bf e0 = load_b_frag(kbase, C3, lane);
    v16bf e1 = load_b_frag(kbase + 32, C3, lane);
    int pl = (qt >= 1) ? 1 : 0;
    v16bf g0 = load_b_frag(kbase + (size_t)pl * 16 * C3,      C3, lane);
    v16bf g1 = load_b_frag(kbase + (size_t)pl * 16 * C3 + 32, C3, lane);
    int kt = 0;
    while (kt + 1 <= qt) {
      tile(kt, e0, e1);
      if (kt + 2 <= qt) {
        e0 = load_b_frag(kbase + (size_t)(kt + 2) * 16 * C3,      C3, lane);
        e1 = load_b_frag(kbase + (size_t)(kt + 2) * 16 * C3 + 32, C3, lane);
      }
      tile(kt + 1, g0, g1);
      if (kt + 3 <= qt) {
        g0 = load_b_frag(kbase + (size_t)(kt + 3) * 16 * C3,      C3, lane);
        g1 = load_b_frag(kbase + (size_t)(kt + 3) * 16 * C3 + 32, C3, lane);
      }
      kt += 2;
    }
    if (kt <= qt) tile(kt, e0, e1);
  }
  float rinv[8];
#pragma unroll
  for (int e = 0; e < 8; e++) {
    float v = rsum[e];
#pragma unroll
    for (int m = 1; m < 16; m <<= 1) v += __shfl_xor(v, m, 32);
    rinv[e] = 1.0f / v;
  }

  // ---- pass 3: write normalized weights + accumulate O = P @ V (32 keys per chunk)
  v8f o[4] = { {0,0,0,0,0,0,0,0}, {0,0,0,0,0,0,0,0}, {0,0,0,0,0,0,0,0}, {0,0,0,0,0,0,0,0} };
  int nch = (q0 + 47) >> 5;                    // 32-key chunks covering causal span
  {
    auto loadK4 = [&](int k0c, v16bf* kf) {
      kf[0] = load_b_frag(kbase + (size_t)k0c * C3,             C3, lane);
      kf[1] = load_b_frag(kbase + (size_t)k0c * C3 + 32,        C3, lane);
      kf[2] = load_b_frag(kbase + (size_t)(k0c + 16) * C3,      C3, lane);
      kf[3] = load_b_frag(kbase + (size_t)(k0c + 16) * C3 + 32, C3, lane);
    };
    auto body = [&](int c, v16bf* kf) {
      int k0c = c * 32;
#pragma unroll
      for (int u = 0; u < 2; u++) {
        int k0 = k0c + u * 16;
        v8f s = {0,0,0,0,0,0,0,0};
        s = WMMA_BF16(qf0, kf[2 * u], s);
        s = WMMA_BF16(qf1, kf[2 * u + 1], s);
#pragma unroll
        for (int e = 0; e < 8; e++) {
          float p = 0.f;
          if (k0 + nn <= q0 + e + moff)
            p = __expf(s[e] * scale - rmax[e]) * rinv[e];
          aw[(e + moff) * T + k0 + nn] = p;                 // normalized weight -> d_out
          pst[(e + moff) * 32 + u * 16 + nn] = f2bf(p);     // stage for A-fragment
        }
      }
      asm volatile("s_wait_dscnt 0" ::: "memory");          // cross-lane LDS visibility (wave32)
      v16bf pf = load_a_frag(pst, 32, lane);
#pragma unroll
      for (int j = 0; j < 4; j++) {
        v16bf vb = load_b_frag(vtb + (size_t)(j * 16) * T + k0c, T, lane);
        o[j] = WMMA_BF16(pf, vb, o[j]);
      }
    };
    v16bf ka[4], kb[4];
    loadK4(0, ka);
    loadK4((nch > 1) ? 32 : 0, kb);            // prologue buf1 (dummy chunk 0 if nch==1)
    int c = 0;
    while (c + 1 < nch) {
      body(c, ka);
      if (c + 2 < nch) loadK4((c + 2) * 32, ka);
      body(c + 1, kb);
      if (c + 3 < nch) loadK4((c + 3) * 32, kb);
      c += 2;
    }
    if (c < nch) body(c, ka);
  }

  // store O (bf16) into [b*T+t, h*64+d] layout for the output projection
#pragma unroll
  for (int j = 0; j < 4; j++)
#pragma unroll
    for (int e = 0; e < 8; e++)
      oh[(size_t)(b * T + q0 + e + moff) * 1024 + h * 64 + j * 16 + nn] = f2bf(o[j][e]);

  // zero-fill strictly-causal-masked columns beyond the computed chunks
  int zstart = nch * 32;
  for (int m = 0; m < 16; m++) {
    float* row = aw + (size_t)m * T;
    for (int col = zstart + lane; col < T; col += 32) row[col] = 0.f;
  }
}

// ---------------------------------------------------------------- launcher
extern "C" void kernel_launch(void* const* d_in, const int* in_sizes, int n_in,
                              void* d_out, int out_size, void* d_ws, size_t ws_size,
                              hipStream_t stream) {
  const float* x     = (const float*)d_in[0];
  const float* w_qkv = (const float*)d_in[1];
  const float* b_qkv = (const float*)d_in[2];
  const float* w_out = (const float*)d_in[3];
  const float* b_out = (const float*)d_in[4];

  const int B = 4, T = 2048, C = 1024, H = 16, DK = 64;
  const int M  = B * T;       // 8192
  const int N3 = 3 * C;       // 3072

  char* ws = (char*)d_ws;
  unsigned short* xb    = (unsigned short*)ws; ws += (size_t)M  * C  * 2;
  unsigned short* wqkvb = (unsigned short*)ws; ws += (size_t)N3 * C  * 2;
  unsigned short* woutb = (unsigned short*)ws; ws += (size_t)C  * C  * 2;
  unsigned short* qkvb  = (unsigned short*)ws; ws += (size_t)M  * N3 * 2;
  unsigned short* vt    = (unsigned short*)ws; ws += (size_t)B * H * DK * T * 2;
  unsigned short* oh    = (unsigned short*)ws; ws += (size_t)M  * C  * 2;

  float* out   = (float*)d_out;
  float* attnw = out + (size_t)M * C;

  f32_to_bf16_k<<<(M * C + 255) / 256, 256, 0, stream>>>(x, xb, M * C);
  f32_to_bf16_k<<<(N3 * C + 255) / 256, 256, 0, stream>>>(w_qkv, wqkvb, N3 * C);
  f32_to_bf16_k<<<(C * C + 255) / 256, 256, 0, stream>>>(w_out, woutb, C * C);

  // QKV projection: [M,N3] = x @ w_qkv^T + b_qkv
  gemm_bf16out<<<((M / 16) * (N3 / 64)) / 8, 256, 0, stream>>>(xb, wqkvb, b_qkv, qkvb, C, N3);

  // V^T repack
  build_vt_k<<<(B * H * DK * T) / 256, 256, 0, stream>>>(qkvb, vt);

  // Fused causal attention (writes attn weights to d_out + O to oh)
  attn_kernel<<<(B * H * (T / 16)) / 8, 256, 0, stream>>>(qkvb, vt, attnw, oh);

  // Output projection: [M,C] = oh @ w_out^T + b_out
  gemm_f32out<<<((M / 16) * (C / 64)) / 8, 256, 0, stream>>>(oh, woutb, b_out, out, C, C);

  (void)in_sizes; (void)n_in; (void)out_size; (void)ws_size;
}
